// SABlock_88897233092934
// MI455X (gfx1250) — matrix operand order repaired and checked by reading, not compiled, not run on hardware
//
#include <hip/hip_runtime.h>
#include <hip/hip_bf16.h>

#define DEVI __device__ __forceinline__

typedef __attribute__((ext_vector_type(16))) __bf16 v16bf;
typedef __attribute__((ext_vector_type(8)))  float  v8f;

// Problem constants
constexpr int NHEAD = 12;
constexpr int DH    = 64;
constexpr int CIN   = 768;        // hidden
constexpr int QKVN  = 3 * CIN;    // 2304
constexpr int NTOK  = 1024;       // 32*32 tokens per image
constexpr int BATCH = 4;
constexpr int BH    = BATCH * NHEAD;  // 48
constexpr int BN    = BATCH * NTOK;   // 4096

DEVI v8f wmma_bf16(v16bf a, v16bf b, v8f c) {
  return __builtin_amdgcn_wmma_f32_16x16x32_bf16(
      /*neg_a=*/false, a, /*neg_b=*/false, b,
      /*c_mod=*/(short)0, c, /*reuse_a=*/false, /*reuse_b=*/false);
}

// A-matrix (16x32 bf16) slot -> K index, contiguous order matching B fragments.
// Slot s of lane half g -> K = g*16 + s.
DEVI int a_slot_k(int s, int g) {
  return g * 16 + s;
}

// ---------------------------------------------------------------------------
// Kernel 1: QKV projection  X(4096x768) @ Wqkv(768x2304)
// Block tile 128x64 (8 waves x 16 rows), K-chunks of 32, bf16 WMMA.
// Weight tile staged TRANSPOSED in LDS so B fragments are contiguous 32B loads.
// q,k stored [bh][token][d] bf16; v stored transposed [bh][d][token] bf16.
// ---------------------------------------------------------------------------
__global__ __launch_bounds__(256)
void qkv_gemm_kernel(const float* __restrict__ x, const float* __restrict__ w,
                     __bf16* __restrict__ qb, __bf16* __restrict__ kb,
                     __bf16* __restrict__ vbT) {
  constexpr int LDA  = 40;  // 32 + 8 pad (bf16 units)
  constexpr int LDBT = 48;  // 32 + 16 pad; 48*2B = 96B row stride (32B-aligned frags)
  __shared__ __attribute__((aligned(32))) __bf16 sA [128 * LDA];
  __shared__ __attribute__((aligned(32))) __bf16 sBT[64 * LDBT];   // [n][k]

  const int t = threadIdx.x;
  const int wave = t >> 5, lane = t & 31, g = lane >> 4, ln = lane & 15;
  const int mtile = blockIdx.y * 128;
  const int ntile = blockIdx.x * 64;

  v8f acc[4] = {};

  for (int kk = 0; kk < CIN; kk += 32) {
    // stage A: 128x32 f32 -> bf16 ; 16 elements per thread
    {
      const int row = t >> 1, col = (t & 1) * 16;
      const float* src = x + (size_t)(mtile + row) * CIN + kk + col;
      __bf16* dst = &sA[row * LDA + col];
      __builtin_prefetch(src + 64, 0, 0);
#pragma unroll
      for (int i = 0; i < 16; i += 4) {
        float4 v = *(const float4*)(src + i);
        dst[i + 0] = (__bf16)v.x; dst[i + 1] = (__bf16)v.y;
        dst[i + 2] = (__bf16)v.z; dst[i + 3] = (__bf16)v.w;
      }
    }
    // stage B transposed: read W rows coalesced, scatter bf16 into sBT[n][k]
    {
      const int row = t >> 3, col = (t & 7) * 8;   // k-row 0..31, n-col base
      const float* src = w + (size_t)(kk + row) * QKVN + ntile + col;
      float4 v0 = *(const float4*)(src + 0);
      float4 v1 = *(const float4*)(src + 4);
      sBT[(col + 0) * LDBT + row] = (__bf16)v0.x;
      sBT[(col + 1) * LDBT + row] = (__bf16)v0.y;
      sBT[(col + 2) * LDBT + row] = (__bf16)v0.z;
      sBT[(col + 3) * LDBT + row] = (__bf16)v0.w;
      sBT[(col + 4) * LDBT + row] = (__bf16)v1.x;
      sBT[(col + 5) * LDBT + row] = (__bf16)v1.y;
      sBT[(col + 6) * LDBT + row] = (__bf16)v1.z;
      sBT[(col + 7) * LDBT + row] = (__bf16)v1.w;
    }
    __syncthreads();

    v16bf a;
#pragma unroll
    for (int s = 0; s < 16; ++s)
      a[s] = sA[(wave * 16 + ln) * LDA + a_slot_k(s, g)];
#pragma unroll
    for (int j = 0; j < 4; ++j) {
      v16bf b = *(const v16bf*)&sBT[(j * 16 + ln) * LDBT + g * 16];
      acc[j] = wmma_bf16(a, b, acc[j]);
    }
    __syncthreads();
  }

  // epilogue: this N-tile belongs to exactly one (which, head)
  const int which = ntile / CIN;             // 0=q 1=k 2=v
  const int head  = (ntile % CIN) / DH;      // 0..11
#pragma unroll
  for (int j = 0; j < 4; ++j) {
#pragma unroll
    for (int r = 0; r < 8; ++r) {
      const int row = mtile + wave * 16 + r + 8 * g;   // global token row
      const int bidx = row >> 10, n = row & (NTOK - 1);
      const int bh = bidx * NHEAD + head;
      const int d = j * 16 + ln;
      const __bf16 val = (__bf16)acc[j][r];
      if (which == 0)      qb[((size_t)bh * NTOK + n) * DH + d] = val;
      else if (which == 1) kb[((size_t)bh * NTOK + n) * DH + d] = val;
      else                 vbT[((size_t)bh * DH + d) * NTOK + n] = val;
    }
  }
}

// ---------------------------------------------------------------------------
// Kernel 2: decomposed relative position bias precompute (tiny VALU kernel)
// relh[bh][n][hk] = sum_c q[bh][n][c] * rel_pos_h[hi - hk + 31][c]
// relw[bh][n][wk] = sum_c q[bh][n][c] * rel_pos_w[wi - wk + 31][c]
// ---------------------------------------------------------------------------
__global__ __launch_bounds__(64)
void rel_bias_kernel(const __bf16* __restrict__ qb,
                     const float* __restrict__ rph, const float* __restrict__ rpw,
                     float* __restrict__ relh, float* __restrict__ relw) {
  const int blk = blockIdx.x;              // [0, 48*1024)
  const int bh = blk >> 10, n = blk & (NTOK - 1);
  const int hi = n >> 5, wi = n & 31;
  const int t = threadIdx.x;               // 64: [0,32)->relh, [32,64)->relw
  const __bf16* qrow = qb + ((size_t)bh * NTOK + n) * DH;
  float acc = 0.f;
  if (t < 32) {
    const float* tab = rph + (size_t)(hi - t + 31) * DH;
#pragma unroll 8
    for (int c = 0; c < DH; ++c) acc += (float)qrow[c] * tab[c];
    relh[(size_t)blk * 32 + t] = acc;
  } else {
    const int wk = t - 32;
    const float* tab = rpw + (size_t)(wi - wk + 31) * DH;
#pragma unroll 8
    for (int c = 0; c < DH; ++c) acc += (float)qrow[c] * tab[c];
    relw[(size_t)blk * 32 + wk] = acc;
  }
}

// ---------------------------------------------------------------------------
// Kernel 3: fused flash attention with decomposed rel-pos bias.
// Grid: (8 query tiles of 128) x (48 head-batches). 256 threads = 8 waves,
// each wave owns 16 query rows. Key loop over 16 tiles of 64 keys.
// ---------------------------------------------------------------------------
__global__ __launch_bounds__(256)
void attn_kernel(const __bf16* __restrict__ qb, const __bf16* __restrict__ kb,
                 const __bf16* __restrict__ vbT,
                 const float* __restrict__ relh, const float* __restrict__ relw,
                 __bf16* __restrict__ aout) {
  constexpr int LDK = 80;   // 64 + 16 pad, multiple of 16 for 32B-aligned v16bf
  constexpr int LDP = 80;
  __shared__ __attribute__((aligned(32))) __bf16 sK [64 * LDK];    // [key][d]
  __shared__ __attribute__((aligned(32))) __bf16 sVT[64 * LDK];    // [d][key]
  __shared__ __attribute__((aligned(32))) __bf16 sP [128 * LDP];   // P tile
  __shared__ float  sRh[128 * 32];    // relh rows for this query tile
  __shared__ float  sRw[128 * 32];

  const int bh = blockIdx.y;
  const int qt = blockIdx.x;
  const int t = threadIdx.x;
  const int wave = t >> 5, lane = t & 31, g = lane >> 4, ln = lane & 15;
  const int qbase = qt * 128;

  // stage rel-pos bias rows once (128x32 f32 each)
  {
    const int i = t * 16;
    const float* srch = relh + ((size_t)bh * NTOK + qbase) * 32 + i;
    const float* srcw = relw + ((size_t)bh * NTOK + qbase) * 32 + i;
#pragma unroll
    for (int u = 0; u < 16; u += 4) {
      *(float4*)&sRh[i + u] = *(const float4*)&srch[u];
      *(float4*)&sRw[i + u] = *(const float4*)&srcw[u];
    }
  }

  // Q A-fragments held in registers for the whole key loop
  v16bf aq0, aq1;
  {
    const __bf16* qsrc = qb + ((size_t)bh * NTOK + qbase + wave * 16 + ln) * DH;
#pragma unroll
    for (int s = 0; s < 16; ++s) {
      aq0[s] = qsrc[a_slot_k(s, g)];
      aq1[s] = qsrc[32 + a_slot_k(s, g)];
    }
  }

  float m_arr[8], l_arr[8];
  v8f o[4] = {};
#pragma unroll
  for (int r = 0; r < 8; ++r) { m_arr[r] = -1e30f; l_arr[r] = 0.f; }

  __syncthreads();  // rel bias staged

  for (int kt = 0; kt < 16; ++kt) {
    // stage K tile [key][d] and V^T tile [d][key], 64x64 bf16 each
    {
      const int row = t >> 2, col = (t & 3) * 16;
      const __bf16* srck = kb + ((size_t)bh * NTOK + kt * 64 + row) * DH + col;
      *(uint4*)&sK[row * LDK + col + 0] = *(const uint4*)&srck[0];
      *(uint4*)&sK[row * LDK + col + 8] = *(const uint4*)&srck[8];
      const __bf16* srcv = vbT + ((size_t)bh * DH + row) * NTOK + kt * 64 + col;
      *(uint4*)&sVT[row * LDK + col + 0] = *(const uint4*)&srcv[0];
      *(uint4*)&sVT[row * LDK + col + 8] = *(const uint4*)&srcv[8];
    }
    __syncthreads();

    // scores S = Q K^T for this wave's 16 rows x 64 keys
    v8f sc[4];
#pragma unroll
    for (int j = 0; j < 4; ++j) {
      const int key = j * 16 + ln;
      v16bf b0 = *(const v16bf*)&sK[key * LDK + 0  + g * 16];
      v16bf b1 = *(const v16bf*)&sK[key * LDK + 32 + g * 16];
      v8f z = {};
      z = wmma_bf16(aq0, b0, z);
      z = wmma_bf16(aq1, b1, z);
      sc[j] = z;
    }

    // scale + decomposed rel-pos bias
    const int hk0 = 2 * kt;
#pragma unroll
    for (int j = 0; j < 4; ++j) {
      const int keyl = j * 16 + ln;
      const int hk = hk0 + (keyl >> 5);
      const int wk = keyl & 31;
#pragma unroll
      for (int r = 0; r < 8; ++r) {
        const int ql = wave * 16 + r + 8 * g;
        const float bias = sRh[ql * 32 + hk] + sRw[ql * 32 + wk];
        sc[j][r] = sc[j][r] * 0.125f + bias;
      }
    }

    // online softmax: rows live across lanes 0-15 / 16-31 (wave32 halves)
#pragma unroll
    for (int r = 0; r < 8; ++r) {
      float mx = fmaxf(fmaxf(sc[0][r], sc[1][r]), fmaxf(sc[2][r], sc[3][r]));
#pragma unroll
      for (int msk = 1; msk < 16; msk <<= 1)
        mx = fmaxf(mx, __shfl_xor(mx, msk, 32));
      const float mnew = fmaxf(m_arr[r], mx);
      const float alpha = __expf(m_arr[r] - mnew);
      m_arr[r] = mnew;
      float psum = 0.f;
#pragma unroll
      for (int j = 0; j < 4; ++j) {
        const float p = __expf(sc[j][r] - mnew);
        sc[j][r] = p;
        psum += p;
      }
#pragma unroll
      for (int msk = 1; msk < 16; msk <<= 1)
        psum += __shfl_xor(psum, msk, 32);
      l_arr[r] = l_arr[r] * alpha + psum;
#pragma unroll
      for (int j = 0; j < 4; ++j) o[j][r] *= alpha;
    }

    // P -> LDS (C-layout -> A-layout relayout; rows are wave-private)
#pragma unroll
    for (int j = 0; j < 4; ++j)
#pragma unroll
      for (int r = 0; r < 8; ++r)
        sP[(wave * 16 + r + 8 * g) * LDP + j * 16 + ln] = (__bf16)sc[j][r];
    asm volatile("s_wait_dscnt 0" ::: "memory");

    v16bf ap0, ap1;
    {
      const int base = (wave * 16 + ln) * LDP;
#pragma unroll
      for (int s = 0; s < 16; ++s) {
        ap0[s] = sP[base + a_slot_k(s, g)];
        ap1[s] = sP[base + 32 + a_slot_k(s, g)];
      }
    }
    // O += P V  (V^T in LDS gives contiguous B fragments)
#pragma unroll
    for (int j = 0; j < 4; ++j) {
      const int dh = j * 16 + ln;
      v16bf bv0 = *(const v16bf*)&sVT[dh * LDK + 0  + g * 16];
      v16bf bv1 = *(const v16bf*)&sVT[dh * LDK + 32 + g * 16];
      o[j] = wmma_bf16(ap0, bv0, o[j]);
      o[j] = wmma_bf16(ap1, bv1, o[j]);
    }
    __syncthreads();  // protect sK/sVT before restage
  }

  // finalize: O /= l, write bf16 to (B, N, head*64+d) layout
  const int bidx = bh / NHEAD, head = bh % NHEAD;
#pragma unroll
  for (int j = 0; j < 4; ++j) {
#pragma unroll
    for (int r = 0; r < 8; ++r) {
      const int ql = wave * 16 + r + 8 * g;
      const int n = qbase + ql;
      const float val = o[j][r] / l_arr[r];
      aout[((size_t)bidx * NTOK + n) * CIN + head * DH + j * 16 + ln] = (__bf16)val;
    }
  }
}

// ---------------------------------------------------------------------------
// Kernel 4: output projection  AO(4096x768 bf16) @ Wout(768x768) + b -> f32
// ---------------------------------------------------------------------------
__global__ __launch_bounds__(256)
void out_proj_kernel(const __bf16* __restrict__ a_in, const float* __restrict__ w,
                     const float* __restrict__ bias, float* __restrict__ out) {
  constexpr int LDA  = 40;
  constexpr int LDBT = 48;
  __shared__ __attribute__((aligned(32))) __bf16 sA [128 * LDA];
  __shared__ __attribute__((aligned(32))) __bf16 sBT[64 * LDBT];   // [n][k]

  const int t = threadIdx.x;
  const int wave = t >> 5, lane = t & 31, g = lane >> 4, ln = lane & 15;
  const int mtile = blockIdx.y * 128;
  const int ntile = blockIdx.x * 64;

  v8f acc[4] = {};

  for (int kk = 0; kk < CIN; kk += 32) {
    {  // stage A (already bf16): 32B per thread
      const int row = t >> 1, col = (t & 1) * 16;
      const __bf16* src = a_in + (size_t)(mtile + row) * CIN + kk + col;
      *(uint4*)&sA[row * LDA + col + 0] = *(const uint4*)&src[0];
      *(uint4*)&sA[row * LDA + col + 8] = *(const uint4*)&src[8];
    }
    {  // stage B transposed with f32->bf16 convert
      const int row = t >> 3, col = (t & 7) * 8;
      const float* src = w + (size_t)(kk + row) * CIN + ntile + col;
      float4 v0 = *(const float4*)(src + 0);
      float4 v1 = *(const float4*)(src + 4);
      sBT[(col + 0) * LDBT + row] = (__bf16)v0.x;
      sBT[(col + 1) * LDBT + row] = (__bf16)v0.y;
      sBT[(col + 2) * LDBT + row] = (__bf16)v0.z;
      sBT[(col + 3) * LDBT + row] = (__bf16)v0.w;
      sBT[(col + 4) * LDBT + row] = (__bf16)v1.x;
      sBT[(col + 5) * LDBT + row] = (__bf16)v1.y;
      sBT[(col + 6) * LDBT + row] = (__bf16)v1.z;
      sBT[(col + 7) * LDBT + row] = (__bf16)v1.w;
    }
    __syncthreads();

    v16bf a;
#pragma unroll
    for (int s = 0; s < 16; ++s)
      a[s] = sA[(wave * 16 + ln) * LDA + a_slot_k(s, g)];
#pragma unroll
    for (int j = 0; j < 4; ++j) {
      v16bf b = *(const v16bf*)&sBT[(j * 16 + ln) * LDBT + g * 16];
      acc[j] = wmma_bf16(a, b, acc[j]);
    }
    __syncthreads();
  }

#pragma unroll
  for (int j = 0; j < 4; ++j) {
#pragma unroll
    for (int r = 0; r < 8; ++r) {
      const int row = mtile + wave * 16 + r + 8 * g;
      const int col = ntile + j * 16 + ln;
      out[(size_t)row * CIN + col] = acc[j][r] + bias[col];
    }
  }
}

// ---------------------------------------------------------------------------
extern "C" void kernel_launch(void* const* d_in, const int* in_sizes, int n_in,
                              void* d_out, int out_size, void* d_ws, size_t ws_size,
                              hipStream_t stream) {
  const float* x     = (const float*)d_in[0];
  const float* w_qkv = (const float*)d_in[1];
  const float* w_out = (const float*)d_in[2];
  const float* b_out = (const float*)d_in[3];
  const float* rph   = (const float*)d_in[4];
  const float* rpw   = (const float*)d_in[5];

  char* ws = (char*)d_ws;
  size_t off = 0;
  auto alloc = [&](size_t bytes) -> void* {
    void* p = ws + off;
    off += (bytes + 255) & ~(size_t)255;
    return p;
  };
  __bf16* qb   = (__bf16*)alloc((size_t)BH * NTOK * DH * 2);  // 6 MB
  __bf16* kb   = (__bf16*)alloc((size_t)BH * NTOK * DH * 2);  // 6 MB
  __bf16* vbT  = (__bf16*)alloc((size_t)BH * DH * NTOK * 2);  // 6 MB (transposed)
  float*  relh = (float*) alloc((size_t)BH * NTOK * 32 * 4);  // 6.3 MB
  float*  relw = (float*) alloc((size_t)BH * NTOK * 32 * 4);  // 6.3 MB
  __bf16* aout = (__bf16*)alloc((size_t)BN * CIN * 2);        // 6.3 MB
  (void)ws_size; (void)in_sizes; (void)n_in; (void)out_size;

  dim3 b256(256);
  qkv_gemm_kernel<<<dim3(QKVN / 64, BN / 128), b256, 0, stream>>>(x, w_qkv, qb, kb, vbT);
  rel_bias_kernel<<<dim3(BH * NTOK), dim3(64), 0, stream>>>(qb, rph, rpw, relh, relw);
  attn_kernel<<<dim3(NTOK / 128, BH), b256, 0, stream>>>(qb, kb, vbT, relh, relw, aout);
  out_proj_kernel<<<dim3(CIN / 64, BN / 128), b256, 0, stream>>>(aout, w_out, b_out, (float*)d_out);
}
